// NormalComparisonModel_29360396435691
// MI455X (gfx1250) — matrix operand order repaired and checked by reading, not compiled, not run on hardware
//
#include <hip/hip_runtime.h>
#include <hip/hip_bf16.h>
#include <math.h>

// ---------- types ----------
typedef __attribute__((ext_vector_type(16))) __bf16 v16bf;
typedef __attribute__((ext_vector_type(8)))  float  v8f;

union Frag16 {
    uint4 u4[2];
    v16bf v;
};

#define D_FEAT   256
#define DENSE    1024
#define HALF_D   512
#define NPAIRS   65536      // 256*256
#define ROWS     64         // pair-rows per block in layer2 kernel
#define HSTRIDE  1032       // 1024 + 8 bf16 elements: pad to kill LDS bank conflicts

static __device__ __forceinline__ unsigned short f2bf(float f) {
    unsigned int u = __float_as_uint(f);
    unsigned int r = (u + 0x7FFFu + ((u >> 16) & 1u)) >> 16;   // RNE
    return (unsigned short)r;
}
static __device__ __forceinline__ float gelu_exact(float x) {
    return 0.5f * x * (1.0f + erff(x * 0.70710678118654752f));
}

// ---------- kernel 1: mean-pool over L=128 ----------
__global__ void pool_kernel(const float* __restrict__ f, float* __restrict__ p) {
    int i = blockIdx.x;          // token row, 0..255
    int k = threadIdx.x;         // feature, 0..255
    const float* base = f + (size_t)i * 128 * D_FEAT + k;
    float s = 0.f;
    #pragma unroll 8
    for (int l = 0; l < 128; ++l) s += base[(size_t)l * D_FEAT];
    p[i * D_FEAT + k] = s * (1.0f / 128.0f);
}

// ---------- kernel 2: W2 -> bf16, transposed to [N=512][K=1024] ----------
__global__ void w2t_kernel(const float* __restrict__ W2, unsigned short* __restrict__ W2T) {
    int idx = blockIdx.x * 256 + threadIdx.x;    // 0 .. 1024*512-1
    int k = idx >> 9;            // 0..1023
    int c = idx & 511;           // 0..511
    W2T[c * DENSE + k] = f2bf(W2[idx]);
}

// ---------- kernel 3: U = p1@(W1a+W1c)+b1  ;  V = p2@(W1b-W1c) ----------
// grid (4, 16, 2) block 256.  z==0 -> U, z==1 -> V
__global__ void layer1_kernel(const float* __restrict__ p1, const float* __restrict__ p2,
                              const float* __restrict__ W1, const float* __restrict__ b1,
                              float* __restrict__ U, float* __restrict__ V) {
    int n  = blockIdx.x * 256 + threadIdx.x;   // 0..1023
    int r0 = blockIdx.y * 16;                  // row group
    bool isV = (blockIdx.z != 0);
    const float* P  = isV ? p2 : p1;
    float acc[16];
    #pragma unroll
    for (int r = 0; r < 16; ++r) acc[r] = 0.f;

    int rowA = isV ? 256 : 0;     // W1b vs W1a block of W1 rows
    for (int k = 0; k < 256; ++k) {
        float wa = W1[(size_t)(rowA + k) * DENSE + n];
        float wc = W1[(size_t)(512  + k) * DENSE + n];
        float w  = isV ? (wa - wc) : (wa + wc);
        #pragma unroll
        for (int r = 0; r < 16; ++r)
            acc[r] += P[(r0 + r) * D_FEAT + k] * w;
    }
    float bb = isV ? 0.f : b1[n];
    float* O = isV ? V : U;
    #pragma unroll
    for (int r = 0; r < 16; ++r)
        O[(size_t)(r0 + r) * DENSE + n] = acc[r] + bb;
}

// ---------- kernel 4: fused layer2 GEMM (WMMA bf16) + gelu + layer3 dot + sigmoid ----------
// Each block: 64 pair-rows x full N=512, K=1024.  8 waves; wave handles 4 column tiles.
__global__ __launch_bounds__(256)
void layer2_kernel(const float* __restrict__ U, const float* __restrict__ V,
                   const float* __restrict__ b2, const float* __restrict__ W3,
                   const float* __restrict__ b3, const unsigned short* __restrict__ W2T,
                   float* __restrict__ out) {
    __shared__ unsigned short h[ROWS * HSTRIDE];   // bf16 activations, padded stride
    __shared__ float rowsum[ROWS];

    const int t    = threadIdx.x;
    const int base = blockIdx.x * ROWS;            // pair index base (64 aligned -> i uniform)
    const int i    = base >> 8;                    // f1 token (uniform per block)
    const int j0   = base & 255;                   // f2 token base
    const float* Urow = U + (size_t)i * DENSE;

    // ---- phase 1: h[m,k] = gelu(U[i,k] + V[j0+m,k]) in bf16 ----
    for (int idx = t; idx < ROWS * DENSE; idx += 256) {
        int m = idx >> 10;
        int k = idx & (DENSE - 1);
        float x = Urow[k] + V[(size_t)(j0 + m) * DENSE + k];
        h[m * HSTRIDE + k] = f2bf(gelu_exact(x));
    }
    if (t < ROWS) rowsum[t] = 0.f;
    __syncthreads();

    // ---- phase 2: WMMA GEMM over K=1024 ----
    const int wave = t >> 5;
    const int lane = t & 31;
    const int half = lane >> 4;     // 0 or 1
    const int lc   = lane & 15;     // column within 16-wide tile / row within tile

    #pragma unroll
    for (int cti = 0; cti < 4; ++cti) {
        const int ct = wave + cti * 8;         // 0..31 column tile
        const int c  = ct * 16 + lc;           // global output column 0..511
        const unsigned short* Bp = W2T + (size_t)c * DENSE + half * 16;

        v8f acc[4];
        #pragma unroll
        for (int rt = 0; rt < 4; ++rt) acc[rt] = (v8f){0.f,0.f,0.f,0.f,0.f,0.f,0.f,0.f};

        for (int k0 = 0; k0 < DENSE; k0 += 32) {
            // B fragment: 16 consecutive K values for this lane's column
            Frag16 bf;
            const uint4* bp4 = reinterpret_cast<const uint4*>(Bp + k0);
            bf.u4[0] = bp4[0];
            bf.u4[1] = bp4[1];

            #pragma unroll
            for (int rt = 0; rt < 4; ++rt) {
                // A fragment from LDS: row = rt*16+lc ; K = k0+half*16..+15
                Frag16 af;
                const uint4* ap4 = reinterpret_cast<const uint4*>(
                    &h[(rt * 16 + lc) * HSTRIDE + k0 + half * 16]);
                af.u4[0] = ap4[0];
                af.u4[1] = ap4[1];   // +8 bf16 elements = +1 uint4
                acc[rt] = __builtin_amdgcn_wmma_f32_16x16x32_bf16(
                    false, af.v, false, bf.v, (short)0, acc[rt], false, false);
            }
        }

        // ---- epilogue: bias + gelu + dot with W3, reduce across 16 columns ----
        const float bias = b2[c];
        const float w3   = W3[c];
        #pragma unroll
        for (int rt = 0; rt < 4; ++rt) {
            #pragma unroll
            for (int r = 0; r < 8; ++r) {
                float x = acc[rt][r] + bias;
                float pv = gelu_exact(x) * w3;
                // sum over the 16 columns held by lanes sharing `half`
                pv += __shfl_xor(pv, 1, 16);
                pv += __shfl_xor(pv, 2, 16);
                pv += __shfl_xor(pv, 4, 16);
                pv += __shfl_xor(pv, 8, 16);
                if (lc == 0)
                    atomicAdd(&rowsum[rt * 16 + half * 8 + r], pv);  // ds_add_f32
            }
        }
    }
    __syncthreads();

    // ---- phase 3: sigmoid, write out ----
    if (t < ROWS) {
        float s = rowsum[t] + b3[0];
        out[base + t] = 1.0f / (1.0f + expf(-s));
    }
}

// ---------- host ----------
extern "C" void kernel_launch(void* const* d_in, const int* in_sizes, int n_in,
                              void* d_out, int out_size, void* d_ws, size_t ws_size,
                              hipStream_t stream) {
    const float* f1 = (const float*)d_in[0];
    const float* f2 = (const float*)d_in[1];
    const float* W1 = (const float*)d_in[2];
    const float* b1 = (const float*)d_in[3];
    const float* W2 = (const float*)d_in[4];
    const float* b2 = (const float*)d_in[5];
    const float* W3 = (const float*)d_in[6];
    const float* b3 = (const float*)d_in[7];
    float* out = (float*)d_out;

    char* ws = (char*)d_ws;
    float* p1 = (float*)(ws);                               // 256*256*4   = 256KB
    float* p2 = (float*)(ws + 262144);                      // 256KB
    float* U  = (float*)(ws + 524288);                      // 256*1024*4  = 1MB
    float* V  = (float*)(ws + 1572864);                     // 1MB
    unsigned short* W2T = (unsigned short*)(ws + 2621440);  // 512*1024*2  = 1MB

    // mean pooling
    pool_kernel<<<256, 256, 0, stream>>>(f1, p1);
    pool_kernel<<<256, 256, 0, stream>>>(f2, p2);
    // W2 -> bf16 transposed
    w2t_kernel<<<(DENSE * HALF_D) / 256, 256, 0, stream>>>(W2, W2T);
    // collapsed layer-1: U (with b1) and V
    layer1_kernel<<<dim3(4, 16, 2), 256, 0, stream>>>(p1, p2, W1, b1, U, V);
    // fused layer2 (WMMA) + layer3 + sigmoid
    layer2_kernel<<<NPAIRS / ROWS, 256, 0, stream>>>(U, V, b2, W3, b3, W2T, out);
}
